// ImageRadarBEVFusion_38517266710715
// MI455X (gfx1250) — matrix operand order repaired and checked by reading, not compiled, not run on hardware
//
#include <hip/hip_runtime.h>
#include <stdint.h>

// ---------------------------------------------------------------------------
// Types for WMMA fragments (gfx1250, wave32, V_WMMA_F32_16X16X32_BF16)
// ---------------------------------------------------------------------------
typedef __attribute__((ext_vector_type(16))) __bf16       v16bf;
typedef __attribute__((ext_vector_type(8)))  float        v8f;
typedef __attribute__((ext_vector_type(4)))  unsigned int u32x4;
typedef __attribute__((ext_vector_type(4)))  int          i32x4;

union Frag { u32x4 q[2]; v16bf v; };

// ---------------------------------------------------------------------------
// CDNA5 async global->LDS copy (ASYNCcnt path), with sync fallback
// Builtin signature (from hipcc diagnostic): (global int4*, local int4*, imm, imm)
// ---------------------------------------------------------------------------
#if defined(__has_builtin)
#if __has_builtin(__builtin_amdgcn_global_load_async_to_lds_b128)
#define HAVE_ASYNC 1
#endif
#endif
#ifndef HAVE_ASYNC
#define HAVE_ASYNC 0
#endif

typedef __attribute__((address_space(1))) i32x4* glb_v4i_p;
typedef __attribute__((address_space(3))) i32x4* lds_v4i_p;

__device__ __forceinline__ void async_cp16(const unsigned short* g, unsigned short* l) {
#if HAVE_ASYNC
  __builtin_amdgcn_global_load_async_to_lds_b128(
      (glb_v4i_p)g, (lds_v4i_p)l, 0, 0);
#else
  *(u32x4*)l = *(const u32x4*)g;
#endif
}

__device__ __forceinline__ void wait_async() {
#if HAVE_ASYNC
#if __has_builtin(__builtin_amdgcn_s_wait_asynccnt)
  __builtin_amdgcn_s_wait_asynccnt(0);
#else
  asm volatile("s_wait_asynccnt 0x0" ::: "memory");
#endif
#endif
}

// ---------------------------------------------------------------------------
// Problem constants (from the reference)
// ---------------------------------------------------------------------------
constexpr int  BATCH = 2;
constexpr int  BEV   = 180;
constexpr int  NP    = BEV * BEV;      // 32400 pixels
constexpr int  OC    = 384;
constexpr int  IC1   = 256;            // image feature channels
constexpr int  CCAT  = 768;            // radar(384) + image(384)
constexpr int  IMH   = 64;
constexpr int  IMW   = 176;
constexpr float IMG_W = 704.0f;
constexpr float IMG_H = 256.0f;

// ---------------------------------------------------------------------------
// f32 -> bf16 (round to nearest even)
// ---------------------------------------------------------------------------
__device__ __forceinline__ unsigned short f2bf(float f) {
  unsigned int u = __float_as_uint(f);
  unsigned int r = u + 0x7FFFu + ((u >> 16) & 1u);
  return (unsigned short)(r >> 16);
}

// ---------------------------------------------------------------------------
// Weight pack: w[OC][IC][KH][KW] f32  ->  wp[kh*KW+kw][OC][IC] bf16
// ---------------------------------------------------------------------------
__global__ void pack_w(const float* __restrict__ w, unsigned short* __restrict__ wp,
                       int oc_n, int ic_n, int kh_n, int kw_n) {
  int n = oc_n * ic_n * kh_n * kw_n;
  for (int i = blockIdx.x * blockDim.x + threadIdx.x; i < n;
       i += gridDim.x * blockDim.x) {
    int kw = i % kw_n; int t = i / kw_n;
    int kh = t % kh_n; t /= kh_n;
    int ic = t % ic_n; int oc = t / ic_n;
    wp[((size_t)(kh * kw_n + kw) * oc_n + oc) * ic_n + ic] = f2bf(w[i]);
  }
}

// ---------------------------------------------------------------------------
// Fold BN into per-channel scale/bias: sb[layer][2][384]
// ---------------------------------------------------------------------------
__global__ void bn_prep(const float* __restrict__ bn1, const float* __restrict__ bn2,
                        const float* __restrict__ bnf1, const float* __restrict__ bnf2,
                        float* __restrict__ sb) {
  const float* bns[4] = {bn1, bn2, bnf1, bnf2};
  int l = blockIdx.x;          // 0..3
  int i = threadIdx.x;         // 0..383
  const float* bn = bns[l];
  float g = bn[i], b = bn[384 + i], m = bn[768 + i], v = bn[1152 + i];
  float s = g * rsqrtf(v + 1e-5f);
  sb[(l * 2 + 0) * 384 + i] = s;
  sb[(l * 2 + 1) * 384 + i] = b - m * s;
}

// ---------------------------------------------------------------------------
// radar f32 -> bf16, written into channels [0,384) of the concat buffer
// ---------------------------------------------------------------------------
__global__ void cvt_radar(const float* __restrict__ r, unsigned short* __restrict__ xcat) {
  size_t n = (size_t)BATCH * OC * NP;
  for (size_t i = (size_t)blockIdx.x * blockDim.x + threadIdx.x; i < n;
       i += (size_t)gridDim.x * blockDim.x) {
    size_t b  = i / ((size_t)OC * NP);
    size_t cp = i - b * (size_t)OC * NP;
    xcat[b * (size_t)CCAT * NP + cp] = f2bf(r[i]);
  }
}

// ---------------------------------------------------------------------------
// Projection + bilinear sampling + valid-height average.
// One block (64 threads) per (batch, bev pixel); each thread owns 4 channels.
// ---------------------------------------------------------------------------
__global__ void sample_kernel(const float* __restrict__ img,
                              const float* __restrict__ l2i,
                              unsigned short* __restrict__ out) {
  const int bp = blockIdx.x;           // b*NP + p
  const int b  = bp / NP;
  const int p  = bp - b * NP;
  const int tid = threadIdx.x;         // 0..63

  const float gx = -54.0f + 0.6f * (float)(p % BEV) + 0.3f;
  const float gy = -54.0f + 0.6f * (float)(p / BEV) + 0.3f;
  const float* M = l2i + b * 12;
  const float* imgb = img + (size_t)b * IC1 * IMH * IMW;

  float acc[4] = {0.f, 0.f, 0.f, 0.f};
  float cnt = 0.f;
  const float zh[3] = {-1.f, 0.f, 1.f};

  for (int zi = 0; zi < 3; ++zi) {
    float z  = zh[zi];
    float p0 = M[0] * gx + M[1] * gy + M[2]  * z + M[3];
    float p1 = M[4] * gx + M[5] * gy + M[6]  * z + M[7];
    float p2 = M[8] * gx + M[9] * gy + M[10] * z + M[11];
    float depth = fmaxf(p2, 1e-5f);
    float px = p0 / depth, py = p1 / depth;
    bool valid = (px >= 0.f) && (px < IMG_W) && (py >= 0.f) && (py < IMG_H) &&
                 (p2 > 1e-5f);
    if (!valid) continue;
    float nx = px / IMG_W * 2.f - 1.f;
    float ny = py / IMG_H * 2.f - 1.f;
    float ix = ((nx + 1.f) * (float)IMW - 1.f) * 0.5f;
    float iy = ((ny + 1.f) * (float)IMH - 1.f) * 0.5f;
    float x0f = floorf(ix), y0f = floorf(iy);
    int   x0 = (int)x0f,    y0 = (int)y0f;
    float wx1 = ix - x0f, wx0 = 1.f - wx1;
    float wy1 = iy - y0f, wy0 = 1.f - wy1;
    int   xs2[2] = {x0, x0 + 1};
    int   ys2[2] = {y0, y0 + 1};
    float wxs[2] = {wx0, wx1};
    float wys[2] = {wy0, wy1};
    for (int cy = 0; cy < 2; ++cy) {
      for (int cx = 0; cx < 2; ++cx) {
        int xc = xs2[cx], yc = ys2[cy];
        if (xc < 0 || xc >= IMW || yc < 0 || yc >= IMH) continue;
        float w = wxs[cx] * wys[cy];
        const float* src = imgb + yc * IMW + xc;
        #pragma unroll
        for (int k = 0; k < 4; ++k)
          acc[k] += w * src[(size_t)(tid + 64 * k) * (IMH * IMW)];
      }
    }
    cnt += 1.f;
  }
  float inv = 1.f / fmaxf(cnt, 1.f);
  #pragma unroll
  for (int k = 0; k < 4; ++k) {
    int c = tid + 64 * k;
    out[((size_t)b * IC1 + c) * NP + p] = f2bf(acc[k] * inv);
  }
}

// ---------------------------------------------------------------------------
// Implicit-GEMM conv (1x1 or 3x3 pad1) with WMMA bf16 -> f32 accumulate.
// Block: 256 threads (8 waves). Output tile: 128 oc x 128 px.
// Waves 4x2 (oc x px); each wave computes 32 oc x 64 px = 2x4 WMMA frags.
// A (weights [tap][OC][IC]) is double-buffered via async global->LDS copies;
// B (activations [IC][NP]) is staged synchronously with halo guards.
// Epilogue: BN scale/bias + ReLU; bf16 out (intermediate) or f32 out (final).
// ---------------------------------------------------------------------------
template <int TAPS, bool OUTF32>
__global__ __launch_bounds__(256) void conv_wmma(
    const unsigned short* __restrict__ X, int IC,
    const unsigned short* __restrict__ Wp,
    const float* __restrict__ scale, const float* __restrict__ bias,
    void* __restrict__ out, long outBatchStride) {
  // LDS. Row strides padded to 16B multiples (A: 80B/row, B: 272B/row).
  __shared__ __align__(16) unsigned short Als[2][128 * 40];  // 2 x (128 oc x 32 k)
  __shared__ __align__(16) unsigned short Bls[32 * 136];     // 32 k x 128 px

  const int tid    = threadIdx.x;
  const int lane   = tid & 31;
  const int wave   = tid >> 5;
  const int woc    = (wave >> 1) * 32;   // wave oc offset inside 128-tile
  const int wpx    = (wave & 1) * 64;    // wave px offset inside 128-tile
  const int ptile  = blockIdx.x * 128;
  const int octile = blockIdx.y * 128;
  const int b      = blockIdx.z;

  const unsigned short* Xb = X + (size_t)b * IC * NP;

  v8f acc[2][4];
  #pragma unroll
  for (int i = 0; i < 2; ++i)
    #pragma unroll
    for (int j = 0; j < 4; ++j)
      acc[i][j] = (v8f){0.f, 0.f, 0.f, 0.f, 0.f, 0.f, 0.f, 0.f};

  // Per-thread staging coordinates
  const int arow = tid >> 1;            // 0..127 (oc row in tile)
  const int acol = (tid & 1) * 16;      // 0 or 16 (k col)
  const int bk   = tid >> 3;            // 0..31  (k row)
  const int bj   = (tid & 7) * 16;      // 0..112 (px col, 16 per thread)

  const int NIT = (IC / 32) * TAPS;     // pipeline iterations (ic-chunk x tap)

  auto issueA = [&](int it, int buf) {
    const int ic  = (it / TAPS) * 32;
    const int tap = it % TAPS;
    const unsigned short* src =
        Wp + ((size_t)(tap * OC + octile + arow) * IC + ic + acol);
    unsigned short* dst = &Als[buf][arow * 40 + acol];
    async_cp16(src, dst);
    async_cp16(src + 8, dst + 8);
  };

  issueA(0, 0);

  #pragma unroll 1
  for (int it = 0; it < NIT; ++it) {
    const int ic  = (it / TAPS) * 32;
    const int tap = it % TAPS;

    // ---- stage B (activations, shifted for the tap) : 32 x 128 bf16 ----
    {
      const unsigned short* src = Xb + (size_t)(ic + bk) * NP;
      const int p0 = ptile + bj;
      if (TAPS == 1) {
        if (p0 + 15 < NP) {
          async_cp16(src + p0,     &Bls[bk * 136 + bj]);
          async_cp16(src + p0 + 8, &Bls[bk * 136 + bj + 8]);
        } else {
          #pragma unroll
          for (int e = 0; e < 16; ++e) {
            int pe = p0 + e;
            Bls[bk * 136 + bj + e] = (pe < NP) ? src[pe] : (unsigned short)0;
          }
        }
      } else {
        const int ky = tap / 3 - 1;
        const int kx = tap % 3 - 1;
        const int y0 = p0 / BEV;
        const int x0 = p0 - y0 * BEV;
        const int sy0 = y0 + ky;
        // Fast path: whole 16-run inside one bev row, shifted run in-bounds.
        if (x0 <= BEV - 16 && sy0 >= 0 && sy0 < BEV &&
            (x0 + kx) >= 0 && (x0 + 15 + kx) < BEV && (p0 + 15) < NP) {
          const unsigned short* s = src + sy0 * BEV + x0 + kx;
          #pragma unroll
          for (int e = 0; e < 16; ++e) Bls[bk * 136 + bj + e] = s[e];
        } else {
          #pragma unroll
          for (int e = 0; e < 16; ++e) {
            int pe = p0 + e;
            unsigned short v = 0;
            if (pe < NP) {
              int y = pe / BEV;
              int x = pe - y * BEV;
              int sy = y + ky, sx = x + kx;
              if (sy >= 0 && sy < BEV && sx >= 0 && sx < BEV)
                v = src[sy * BEV + sx];
            }
            Bls[bk * 136 + bj + e] = v;
          }
        }
      }
    }

    wait_async();            // A(it) (and async B for TAPS==1) landed in LDS
    __syncthreads();

    if (it + 1 < NIT) issueA(it + 1, (it + 1) & 1);  // overlap with compute

    // ---- fragments (ISA 7.12.2 bf16 layouts) ----
    const unsigned short* Ab = Als[it & 1];
    Frag af[2], bf_[4];
    const int half = lane >> 4;
    #pragma unroll
    for (int fi = 0; fi < 2; ++fi) {
      int row = woc + fi * 16 + (lane & 15);
      af[fi].q[0] = *(const u32x4*)&Ab[row * 40 + half * 8];       // K 8h..8h+7
      af[fi].q[1] = *(const u32x4*)&Ab[row * 40 + 16 + half * 8];  // K 16+8h..
    }
    #pragma unroll
    for (int fj = 0; fj < 4; ++fj) {
      int base = wpx + fj * 16 + (lane & 15);   // lane = N col, regs span K
      unsigned short tb[16];
      #pragma unroll
      for (int s = 0; s < 8; ++s) {
        tb[s]     = Bls[(half * 8 + s) * 136 + base];
        tb[s + 8] = Bls[(16 + half * 8 + s) * 136 + base];
      }
      __builtin_memcpy(&bf_[fj], tb, 32);
    }

    // ---- 8 WMMAs per wave ----
    #pragma unroll
    for (int fi = 0; fi < 2; ++fi)
      #pragma unroll
      for (int fj = 0; fj < 4; ++fj)
        acc[fi][fj] = __builtin_amdgcn_wmma_f32_16x16x32_bf16(
            false, af[fi].v, false, bf_[fj].v, (short)0, acc[fi][fj],
            false, false);
    __syncthreads();
  }

  // ---- epilogue: BN + ReLU, store ----
  #pragma unroll
  for (int fi = 0; fi < 2; ++fi) {
    #pragma unroll
    for (int fj = 0; fj < 4; ++fj) {
      int p = ptile + wpx + fj * 16 + (lane & 15);
      if (p >= NP) continue;
      #pragma unroll
      for (int r = 0; r < 8; ++r) {
        int oc = octile + woc + fi * 16 + (lane >> 4) * 8 + r;
        float v = acc[fi][fj][r] * scale[oc] + bias[oc];
        v = fmaxf(v, 0.f);
        size_t idx = (size_t)b * outBatchStride + (size_t)oc * NP + p;
        if (OUTF32) ((float*)out)[idx] = v;
        else        ((unsigned short*)out)[idx] = f2bf(v);
      }
    }
  }
}

// ---------------------------------------------------------------------------
// Launch
// ---------------------------------------------------------------------------
extern "C" void kernel_launch(void* const* d_in, const int* in_sizes, int n_in,
                              void* d_out, int out_size, void* d_ws, size_t ws_size,
                              hipStream_t stream) {
  (void)in_sizes; (void)n_in; (void)out_size; (void)ws_size;

  const float* radar = (const float*)d_in[0];
  const float* imgf  = (const float*)d_in[1];
  const float* l2i   = (const float*)d_in[2];
  const float* w1    = (const float*)d_in[3];
  const float* bn1   = (const float*)d_in[4];
  const float* w2    = (const float*)d_in[5];
  const float* bn2   = (const float*)d_in[6];
  const float* wf1   = (const float*)d_in[7];
  const float* bnf1  = (const float*)d_in[8];
  const float* wf2   = (const float*)d_in[9];
  const float* bnf2  = (const float*)d_in[10];

  char* ws = (char*)d_ws;
  // workspace layout (bytes); h3 aliases imgbev (+start of h1), both dead by then
  const size_t offImgbev = 0;
  const size_t offH1     = (size_t)BATCH * IC1 * NP * 2;
  const size_t offXcat   = offH1 + (size_t)BATCH * OC * NP * 2;
  const size_t offW1p    = offXcat + (size_t)BATCH * CCAT * NP * 2;
  const size_t offW2p    = offW1p + (size_t)OC * IC1 * 2;
  const size_t offWf1p   = offW2p + (size_t)9 * OC * OC * 2;
  const size_t offWf2p   = offWf1p + (size_t)9 * OC * CCAT * 2;
  const size_t offSb     = offWf2p + (size_t)9 * OC * OC * 2;

  unsigned short* imgbev = (unsigned short*)(ws + offImgbev);
  unsigned short* h1     = (unsigned short*)(ws + offH1);
  unsigned short* xcat   = (unsigned short*)(ws + offXcat);
  unsigned short* h3     = (unsigned short*)(ws + offImgbev);     // alias
  unsigned short* wp1    = (unsigned short*)(ws + offW1p);
  unsigned short* wp2    = (unsigned short*)(ws + offW2p);
  unsigned short* wpf1   = (unsigned short*)(ws + offWf1p);
  unsigned short* wpf2   = (unsigned short*)(ws + offWf2p);
  float*          sb     = (float*)(ws + offSb);

  // 1) pack weights to bf16 [tap][oc][ic]
  pack_w<<<256, 256, 0, stream>>>(w1,  wp1,  OC, IC1,  1, 1);
  pack_w<<<256, 256, 0, stream>>>(w2,  wp2,  OC, OC,   3, 3);
  pack_w<<<512, 256, 0, stream>>>(wf1, wpf1, OC, CCAT, 3, 3);
  pack_w<<<256, 256, 0, stream>>>(wf2, wpf2, OC, OC,   3, 3);

  // 2) fold BN
  bn_prep<<<4, 384, 0, stream>>>(bn1, bn2, bnf1, bnf2, sb);

  // 3) radar -> bf16 into concat channels [0,384)
  cvt_radar<<<4096, 256, 0, stream>>>(radar, xcat);

  // 4) projection + bilinear sampling -> imgbev bf16
  sample_kernel<<<BATCH * NP, 64, 0, stream>>>(imgf, l2i, imgbev);

  // 5..8) conv stack via WMMA implicit GEMM
  dim3 cgrid((NP + 127) / 128, OC / 128, BATCH);   // (254, 3, 2)
  conv_wmma<1, false><<<cgrid, 256, 0, stream>>>(
      imgbev, IC1, wp1, sb + 0 * 384, sb + 1 * 384,
      (void*)h1, (long)OC * NP);
  conv_wmma<9, false><<<cgrid, 256, 0, stream>>>(
      h1, OC, wp2, sb + 2 * 384, sb + 3 * 384,
      (void*)(xcat + (size_t)OC * NP), (long)CCAT * NP);  // concat ch [384,768)
  conv_wmma<9, false><<<cgrid, 256, 0, stream>>>(
      xcat, CCAT, wpf1, sb + 4 * 384, sb + 5 * 384,
      (void*)h3, (long)OC * NP);
  conv_wmma<9, true><<<cgrid, 256, 0, stream>>>(
      h3, OC, wpf2, sb + 6 * 384, sb + 7 * 384,
      d_out, (long)OC * NP);
}
